// Self_Attention_1632087572976
// MI455X (gfx1250) — matrix operand order repaired and checked by reading, not compiled, not run on hardware
//
#include <hip/hip_runtime.h>
#include <hip/hip_bf16.h>

// ---------------------------------------------------------------------------
// SAGAN self-attention, fused flash-attention style, f16 WMMA w/ f32 accum.
// B=4, C=512, Ch=64, N=64*64=4096.
// Stage 0: weights f32 -> f16 (once).
// Stage 1: f/g/h projections (WMMA).
// Stage 2: fused flash attention (WMMA, online softmax, no NxN matrix).
// Stage 3: output projection + residual (WMMA, fused epilogue).
// ---------------------------------------------------------------------------

typedef __attribute__((ext_vector_type(16))) _Float16 v16h;
typedef __attribute__((ext_vector_type(8)))  _Float16 v8h;
typedef __attribute__((ext_vector_type(4)))  _Float16 v4h;
typedef __attribute__((ext_vector_type(8)))  float    v8f;
typedef __attribute__((ext_vector_type(4)))  float    v4f;

#define BATCH 4
#define CIN   512
#define CH    64
#define NPIX  4096

// ---- WMMA fragment helpers (gfx1250 16x16x32 f16, wave32) -----------------
// A (16x32, MxK): lane<16 holds row M=lane, K 0..15; lane>=16 holds
// row M=lane-16, K 16..31. `row32` points at K=0 of a contiguous
// 32-half row.
static __device__ __forceinline__ v16h wmma_ld_a_f16(const _Float16* row32, int hi) {
  const v8h* p = (const v8h*)row32;
  v8h lo = p[2 * hi];      // halfs [hi*16 , hi*16+8)
  v8h up = p[2 * hi + 1];  // halfs [hi*16+8 , hi*16+16)
  return __builtin_shufflevector(lo, up, 0,1,2,3,4,5,6,7,8,9,10,11,12,13,14,15);
}
// A fragment built from an f32 row (convert on the fly; used for LDS-staged x).
static __device__ __forceinline__ v16h wmma_cvt_a_f32(const float* row32, int hi) {
  v16h a;
#pragma unroll
  for (int i = 0; i < 8; ++i) a[i]     = (_Float16)row32[hi * 16 + i];
#pragma unroll
  for (int i = 0; i < 8; ++i) a[8 + i] = (_Float16)row32[hi * 16 + 8 + i];
  return a;
}
// B (32x16, KxN): lane n<16 holds column n, K = 0..15 of the chunk; lane
// n>=16 holds column n-16, K = 16..31.  Caller passes the per-lane pointer
// to 16 contiguous halfs (must be 32B aligned).
static __device__ __forceinline__ v16h wmma_ld_b_f16(const _Float16* col16) {
  return *(const v16h*)col16;
}
static __device__ __forceinline__ v8f wmma_f32_16x16x32(v16h a, v16h b, v8f c) {
  return __builtin_amdgcn_wmma_f32_16x16x32_f16(false, a, false, b, (short)0, c,
                                                false, false);
}
static __device__ __forceinline__ v8f v8f_zero() {
  v8f z = {0.f, 0.f, 0.f, 0.f, 0.f, 0.f, 0.f, 0.f};
  return z;
}

// ---------------------------------------------------------------------------
// Kernel 0: convert Wf/Wg/Wh (64x512 each) and Wv (512x64) to f16, row-major
// unchanged.  One-shot, 131072 elements; removes all weight f32->f16 VALU
// work from the hot GEMM loops (weights are reused by every block).
// ---------------------------------------------------------------------------
__global__ __launch_bounds__(256) void cvt_weights_kernel(
    const float* __restrict__ Wf, const float* __restrict__ Wg,
    const float* __restrict__ Wh, const float* __restrict__ Wv,
    _Float16* __restrict__ w16) {
  const float* srcs[4] = {Wf, Wg, Wh, Wv};
  const int idx4 = blockIdx.x * 256 + threadIdx.x;  // 0..32767, 4 elems each
  const int e    = idx4 * 4;
  const int reg  = e >> 15;        // 32768 elements per weight tensor
  const int off  = e & 32767;
  v4f s = *(const v4f*)(srcs[reg] + off);
  v4h d;
#pragma unroll
  for (int i = 0; i < 4; ++i) d[i] = (_Float16)s[i];
  *(v4h*)(w16 + e) = d;
}

// ---------------------------------------------------------------------------
// Kernel 1: fused f/g/h projections.
//   f = Wf@x+bf  -> fT  [B][N][Ch]  (pixel-major, for Q A-fragments)
//   g = Wg@x+bg  -> gT  [B][N][Ch]  (pixel-major, for K B-fragments)
//   h = Wh@x+bh  -> hM  [B][Ch][N]  (channel-major, for V B-fragments)
// Block = 128 threads (4 waves), handles 64 pixels; wave w owns 16 pixels.
// x chunk (32 in-ch x 64 pix) is staged through LDS (transposed) so the
// A-operand gather reads contiguous rows.  Weights come pre-converted (f16).
// ---------------------------------------------------------------------------
__global__ __launch_bounds__(128) void proj_kernel(
    const float* __restrict__ x,
    const _Float16* __restrict__ Wf16, const float* __restrict__ bf,
    const _Float16* __restrict__ Wg16, const float* __restrict__ bg,
    const _Float16* __restrict__ Wh16, const float* __restrict__ bh,
    _Float16* __restrict__ fT, _Float16* __restrict__ gT,
    _Float16* __restrict__ hM) {
  __shared__ float lx[64 * 33];  // [pixel][in-ch] with pad (gcd(33,64)=1)

  const int t    = threadIdx.x;
  const int w    = t >> 5;
  const int lane = t & 31;
  const int li   = lane & 15;
  const int hi   = lane >> 4;
  const int bb   = blockIdx.y;
  const int p0   = blockIdx.x * 64;

  const _Float16* Ws[3] = {Wf16, Wg16, Wh16};
  const float*    Bs[3] = {bf, bg, bh};

  v8f acc[12];
#pragma unroll
  for (int i = 0; i < 12; ++i) acc[i] = v8f_zero();

  for (int kc = 0; kc < 16; ++kc) {  // 16 chunks of 32 input channels
    const float* xs = x + ((size_t)(bb * CIN + kc * 32) * NPIX) + p0;
#pragma unroll
    for (int i = 0; i < 16; ++i) {   // 2048 f32 cooperative, coalesced load
      int idx = i * 128 + t;
      int c   = idx >> 6;   // in-channel within chunk
      int p   = idx & 63;   // pixel within block tile
      lx[p * 33 + c] = xs[(size_t)c * NPIX + p];
    }
    __syncthreads();
    v16h a = wmma_cvt_a_f32(&lx[(w * 16 + li) * 33], hi);
    __syncthreads();  // protect LDS before next chunk overwrites
#pragma unroll
    for (int s = 0; s < 3; ++s) {
#pragma unroll
      for (int ot = 0; ot < 4; ++ot) {
        const _Float16* wp =
            Ws[s] + (size_t)(ot * 16 + li) * CIN + kc * 32 + hi * 16;
        acc[s * 4 + ot] = wmma_f32_16x16x32(a, wmma_ld_b_f16(wp),
                                            acc[s * 4 + ot]);
      }
    }
  }

  // Epilogue: bias add + store in the layouts the next kernels want.
  const int pixbase = p0 + w * 16 + hi * 8;  // C-layout: row = r + 8*hi
#pragma unroll
  for (int s = 0; s < 3; ++s) {
#pragma unroll
    for (int ot = 0; ot < 4; ++ot) {
      const int   oc   = ot * 16 + li;
      const float bias = Bs[s][oc];
      const v8f   vacc = acc[s * 4 + ot];
      if (s < 2) {  // pixel-major fT / gT
        _Float16* dst = (s == 0 ? fT : gT);
#pragma unroll
        for (int r = 0; r < 8; ++r)
          dst[(size_t)(bb * NPIX + pixbase + r) * CH + oc] =
              (_Float16)(vacc[r] + bias);
      } else {      // channel-major hM, 8 consecutive pixels -> one b128 store
        v8h pk;
#pragma unroll
        for (int r = 0; r < 8; ++r) pk[r] = (_Float16)(vacc[r] + bias);
        *(v8h*)(hM + (size_t)(bb * CH + oc) * NPIX + pixbase) = pk;
      }
    }
  }
}

// ---------------------------------------------------------------------------
// Kernel 2: fused flash attention.
//   O = softmax(Q K^T) V  with Q=f^T, K=g^T, V=h^T, d=Ch=64, no scale.
// Block = 128 threads (4 waves -> 256 workgroups to fill more WGPs). Each
// wave: 16 queries, full d=64 output, streams keys in blocks of 32. Online
// softmax state lives in VGPRs; the 4096x4096 attention matrix is never
// materialized (saves ~0.5 GB of HBM traffic).
// Output hvT [B][N][Ch] (pixel-major) as f16.
// ---------------------------------------------------------------------------
__global__ __launch_bounds__(128) void attn_kernel(
    const _Float16* __restrict__ fT, const _Float16* __restrict__ gT,
    const _Float16* __restrict__ hM, _Float16* __restrict__ hvT) {
  __shared__ _Float16 Pb[4 * 16 * 32];  // per-wave 16x32 prob tile (4 KB)

  const int t    = threadIdx.x;
  const int w    = t >> 5;
  const int lane = t & 31;
  const int li   = lane & 15;
  const int hi   = lane >> 4;
  const int bb   = blockIdx.y;
  const int i0   = blockIdx.x * 64 + w * 16;

  // Q fragments: 16 queries x 64 channels (two K=32 chunks), loaded once.
  const _Float16* qrow = fT + (size_t)(bb * NPIX + i0 + li) * CH;
  const v16h aQ0 = wmma_ld_a_f16(qrow, hi);
  const v16h aQ1 = wmma_ld_a_f16(qrow + 32, hi);

  float mr[8], lr[8];
  v8f   O[4];
#pragma unroll
  for (int r = 0; r < 8; ++r) { mr[r] = -3.0e38f; lr[r] = 0.f; }
#pragma unroll
  for (int dg = 0; dg < 4; ++dg) O[dg] = v8f_zero();

  _Float16* pb = &Pb[w * 512];

  for (int j0 = 0; j0 < NPIX; j0 += 32) {
    // ---- S = Q K^T for 32 keys (two 16-col groups, K=64 chained) ----
    const _Float16* k0 = gT + (size_t)(bb * NPIX + j0 + li) * CH + hi * 16;
    const _Float16* k1 = gT + (size_t)(bb * NPIX + j0 + 16 + li) * CH + hi * 16;
    v8f S0 = v8f_zero(), S1 = v8f_zero();
    S0 = wmma_f32_16x16x32(aQ0, wmma_ld_b_f16(k0),      S0);
    S0 = wmma_f32_16x16x32(aQ1, wmma_ld_b_f16(k0 + 32), S0);
    S1 = wmma_f32_16x16x32(aQ0, wmma_ld_b_f16(k1),      S1);
    S1 = wmma_f32_16x16x32(aQ1, wmma_ld_b_f16(k1 + 32), S1);

    // ---- online softmax across the 32 new columns ----
    float corr[8];
#pragma unroll
    for (int r = 0; r < 8; ++r) {
      float mx = fmaxf(S0[r], S1[r]);
      mx = fmaxf(mx, __shfl_xor(mx, 1, 32));
      mx = fmaxf(mx, __shfl_xor(mx, 2, 32));
      mx = fmaxf(mx, __shfl_xor(mx, 4, 32));
      mx = fmaxf(mx, __shfl_xor(mx, 8, 32));  // row max over 16-lane half
      const float mn = fmaxf(mr[r], mx);
      corr[r] = __expf(mr[r] - mn);
      mr[r]   = mn;
      const float p0v = __expf(S0[r] - mn);
      const float p1v = __expf(S1[r] - mn);
      float rs = p0v + p1v;
      rs += __shfl_xor(rs, 1, 32);
      rs += __shfl_xor(rs, 2, 32);
      rs += __shfl_xor(rs, 4, 32);
      rs += __shfl_xor(rs, 8, 32);
      lr[r] = lr[r] * corr[r] + rs;
      // stash probabilities (row-major 16x32) for the A-operand re-layout
      pb[(r + 8 * hi) * 32 + li]      = (_Float16)p0v;
      pb[(r + 8 * hi) * 32 + li + 16] = (_Float16)p1v;
    }
#pragma unroll
    for (int dg = 0; dg < 4; ++dg)
#pragma unroll
      for (int r = 0; r < 8; ++r) O[dg][r] *= corr[r];

    // ---- O += P (16x32) * V (32x64): 4 WMMA over the d dimension ----
    const v16h aP = wmma_ld_a_f16(&pb[li * 32], hi);
#pragma unroll
    for (int dg = 0; dg < 4; ++dg) {
      const _Float16* vb =
          hM + (size_t)(bb * CH + dg * 16 + li) * NPIX + j0 + hi * 16;
      O[dg] = wmma_f32_16x16x32(aP, wmma_ld_b_f16(vb), O[dg]);
    }
  }

  // ---- normalize and store hvT [B][N][Ch] ----
#pragma unroll
  for (int r = 0; r < 8; ++r) {
    const float inv = 1.0f / lr[r];
    const size_t row = (size_t)(bb * NPIX + i0 + r + 8 * hi) * CH;
#pragma unroll
    for (int dg = 0; dg < 4; ++dg)
      hvT[row + dg * 16 + li] = (_Float16)(O[dg][r] * inv);
  }
}

// ---------------------------------------------------------------------------
// Kernel 3: v = Wv @ hv + bv, out = x + sigma * v (residual fused).
// Computed as v^T = hv^T * Wv^T so D rows = pixels (coalesced f32 stores).
// Block = 128 threads (4 waves, 256 workgroups); each wave: 16 pixels x all
// 512 output channels.  Wv comes pre-converted to f16.
// ---------------------------------------------------------------------------
__global__ __launch_bounds__(128) void outproj_kernel(
    const _Float16* __restrict__ hvT, const _Float16* __restrict__ Wv16,
    const float* __restrict__ bv, const float* __restrict__ x,
    const float* __restrict__ sigma, float* __restrict__ out) {
  const int t    = threadIdx.x;
  const int w    = t >> 5;
  const int lane = t & 31;
  const int li   = lane & 15;
  const int hi   = lane >> 4;
  const int bb   = blockIdx.y;
  const int i0   = blockIdx.x * 64 + w * 16;

  const _Float16* hrow = hvT + (size_t)(bb * NPIX + i0 + li) * CH;
  const v16h aH0 = wmma_ld_a_f16(hrow, hi);
  const v16h aH1 = wmma_ld_a_f16(hrow + 32, hi);
  const float sg = sigma[0];

  for (int ct = 0; ct < 32; ++ct) {  // 32 tiles of 16 output channels
    const int       oc = ct * 16 + li;
    const _Float16* wp = Wv16 + (size_t)oc * CH + hi * 16;  // Wv^T B-operand
    v8f acc = v8f_zero();
    acc = wmma_f32_16x16x32(aH0, wmma_ld_b_f16(wp),      acc);
    acc = wmma_f32_16x16x32(aH1, wmma_ld_b_f16(wp + 32), acc);

    const float bias = bv[oc];
    const size_t base = (size_t)(bb * CIN + oc) * NPIX + i0 + hi * 8;
    v4f xa = *(const v4f*)(x + base);
    v4f xb = *(const v4f*)(x + base + 4);
    v4f oa, ob;
#pragma unroll
    for (int r = 0; r < 4; ++r) {
      oa[r] = xa[r] + sg * (acc[r] + bias);
      ob[r] = xb[r] + sg * (acc[4 + r] + bias);
    }
    *(v4f*)(out + base)     = oa;
    *(v4f*)(out + base + 4) = ob;
  }
}

// ---------------------------------------------------------------------------
extern "C" void kernel_launch(void* const* d_in, const int* in_sizes, int n_in,
                              void* d_out, int out_size, void* d_ws,
                              size_t ws_size, hipStream_t stream) {
  const float* x     = (const float*)d_in[0];
  const float* Wf    = (const float*)d_in[1];
  const float* bf    = (const float*)d_in[2];
  const float* Wg    = (const float*)d_in[3];
  const float* bg    = (const float*)d_in[4];
  const float* Wh    = (const float*)d_in[5];
  const float* bh    = (const float*)d_in[6];
  const float* Wv    = (const float*)d_in[7];
  const float* bv    = (const float*)d_in[8];
  const float* sigma = (const float*)d_in[9];
  float*       out   = (float*)d_out;

  // Workspace layout (f16): fT | gT | hM | hvT (each B*N*Ch = 1Mi halfs, 2 MB)
  // then the converted weights (4 x 32768 halfs = 256 KB).
  const size_t elems = (size_t)BATCH * NPIX * CH;
  _Float16* fT   = (_Float16*)d_ws;
  _Float16* gT   = fT + elems;
  _Float16* hM   = gT + elems;
  _Float16* hvT  = hM + elems;
  _Float16* w16  = hvT + elems;           // Wf16|Wg16|Wh16|Wv16
  _Float16* Wf16 = w16;
  _Float16* Wg16 = w16 + 32768;
  _Float16* Wh16 = w16 + 65536;
  _Float16* Wv16 = w16 + 98304;

  cvt_weights_kernel<<<128, 256, 0, stream>>>(Wf, Wg, Wh, Wv, w16);

  dim3 gProj(NPIX / 64, BATCH);    // 64 pixels / block of 128 threads
  proj_kernel<<<gProj, 128, 0, stream>>>(x, Wf16, bf, Wg16, bg, Wh16, bh,
                                         fT, gT, hM);

  dim3 gAttn(NPIX / 64, BATCH);    // 64 queries / block of 128 threads
  attn_kernel<<<gAttn, 128, 0, stream>>>(fT, gT, hM, hvT);

  dim3 gOut(NPIX / 64, BATCH);     // 64 pixels / block of 128 threads
  outproj_kernel<<<gOut, 128, 0, stream>>>(hvT, Wv16, bv, x, sigma, out);

  (void)in_sizes; (void)n_in; (void)out_size; (void)ws_size;
}
